// VariationalGATEncoder_54597624267030
// MI455X (gfx1250) — matrix operand order repaired and checked by reading, not compiled, not run on hardware
//
#include <hip/hip_runtime.h>
#include <hip/hip_bf16.h>

// ---------------------------------------------------------------------------
// VariationalGATEncoder for MI455X (gfx1250, wave32, WMMA)
//   3x bf16 WMMA GEMM (x@W1, h1@W_mu, h1@W_ls) + segment-softmax attention
//   Weights pre-transposed to [Cout,K] so both A and B WMMA fragments load
//   from LDS as contiguous ds_load_b128 (no scalar u16 gathers).
// ---------------------------------------------------------------------------

typedef __attribute__((ext_vector_type(16))) __bf16 v16bf;
typedef __attribute__((ext_vector_type(8)))  float  v8f;

#define GAT_N 50000
#define GAT_E 800000
#define NEG_SLOPE 0.2f

__device__ __forceinline__ __bf16 f2bf(float f) {
  // round-to-nearest-even fp32 -> bf16 (bit-exact, no reliance on cast codegen)
  unsigned u = __float_as_uint(f);
  unsigned r = (u + 0x7FFFu + ((u >> 16) & 1u)) >> 16;
  union { unsigned short s; __bf16 b; } cv;
  cv.s = (unsigned short)r;
  return cv.b;
}

__device__ __forceinline__ float leaky(float v) {
  return v > 0.f ? v : NEG_SLOPE * v;
}

// -------------------------------- conversions ------------------------------

__global__ void f32_to_bf16_kernel(const float* __restrict__ in,
                                   __bf16* __restrict__ out, long long n) {
  long long i = (long long)blockIdx.x * blockDim.x + threadIdx.x;
  long long stride = (long long)gridDim.x * blockDim.x;
  for (; i < n; i += stride) out[i] = f2bf(in[i]);
}

// in: [R,C] row-major fp32 -> out: [C,R] row-major bf16 (transposed)
__global__ void f32_to_bf16_t_kernel(const float* __restrict__ in,
                                     __bf16* __restrict__ out, int R, int C) {
  int total = R * C;
  int stride = gridDim.x * blockDim.x;
  for (int i = blockIdx.x * blockDim.x + threadIdx.x; i < total; i += stride) {
    int r = i / C, c = i % C;
    out[(long long)c * R + r] = f2bf(in[i]);
  }
}

// ------------------------------- WMMA GEMM ---------------------------------
// D[N,Cout] = A[N,K](bf16,row-major) * BT[Cout,K](bf16,row-major = B^T)
// Block tile 128x128, 8 waves in 4x2 layout, each wave 32x64 (8 wmma accums).
// K stepped by 32 (v_wmma_f32_16x16x32_bf16).

__global__ __launch_bounds__(256)
void gemm_bf16_wmma(const __bf16* __restrict__ A, const __bf16* __restrict__ BT,
                    float* __restrict__ D, int N, int K, int Cout) {
  // stride 40 bf16 = 20 dwords: padded stride -> conflict-free row reads
  __shared__ __bf16 As[128][40];   // [row][k]
  __shared__ __bf16 Bs[128][40];   // [col][k]   (weights pre-transposed)

  const int tid  = threadIdx.x;
  const int lane = tid & 31;
  const int wave = tid >> 5;   // 0..7
  const int wy   = wave >> 1;  // 0..3 (row group of 32)
  const int wx   = wave & 1;   // 0..1 (col group of 64)

  const int rowBase = blockIdx.x * 128;
  const int colBase = blockIdx.y * 128;

  const int l15 = lane & 15;
  const int hi  = lane >> 4;   // half-wave index

  const bool fullTile = (rowBase + 128 <= N);

  v8f acc[2][4];
  {
    v8f z = {};
#pragma unroll
    for (int i = 0; i < 2; ++i)
#pragma unroll
      for (int j = 0; j < 4; ++j) acc[i][j] = z;
  }

  // staging assignments (each thread: 16 bf16 of As and 16 bf16 of Bs)
  const int srow = tid >> 1;           // 0..127
  const int sh   = (tid & 1) << 4;     // 0 or 16
  const int agrow = rowBase + srow;
  const bool aok = fullTile || (agrow < N);
  const __bf16* aptr = A  + (long long)agrow * K + sh;
  const __bf16* bptr = BT + (long long)(colBase + srow) * K + sh;

  for (int k0 = 0; k0 < K; k0 += 32) {
    // ---- stage A tile: 128 rows x 32 k
    {
      uint4 va = {0, 0, 0, 0}, vb = {0, 0, 0, 0};
      if (aok) {
        va = *(const uint4*)(aptr + k0);
        vb = *(const uint4*)(aptr + k0 + 8);
      }
      *(uint4*)&As[srow][sh]     = va;
      *(uint4*)&As[srow][sh + 8] = vb;
    }
    // ---- stage B tile: 128 cols x 32 k (contiguous: BT is [Cout,K])
    {
      uint4 wa = *(const uint4*)(bptr + k0);
      uint4 wb = *(const uint4*)(bptr + k0 + 8);
      *(uint4*)&Bs[srow][sh]     = wa;
      *(uint4*)&Bs[srow][sh + 8] = wb;
    }
    // prefetch next K-slice while this tile is consumed
    if (k0 + 32 < K) {
      __builtin_prefetch(aptr + k0 + 32, 0, 0);
      __builtin_prefetch(bptr + k0 + 32, 0, 0);
    }
    __syncthreads();

    // A fragments (16x32 bf16, ISA 7.12.2):
    //   lanes 0-15 : row M=l15, elems 0..7 -> K=0..7,  8..15 -> K=16..23
    //   lanes16-31 : row M=l15, elems 0..7 -> K=8..15, 8..15 -> K=24..31
    v16bf af[2];
#pragma unroll
    for (int i = 0; i < 2; ++i) {
      int ar = wy * 32 + i * 16 + l15;
#pragma unroll
      for (int j = 0; j < 8; ++j) {
        af[i][j]     = As[ar][hi * 8 + j];
        af[i][j + 8] = As[ar][hi * 8 + 16 + j];
      }
    }
#pragma unroll
    for (int j2 = 0; j2 < 4; ++j2) {
      // B fragment (32x16): element e<8 -> K = hi*8 + e, e>=8 -> K = 16 + hi*8 + (e-8), col N = l15
      // Bs is [col][k]; matches the A fragment k order.
      v16bf bf;
      int bc = wx * 64 + j2 * 16 + l15;
#pragma unroll
      for (int t = 0; t < 16; ++t) bf[t] = Bs[bc][hi * 8 + (t >> 3) * 16 + (t & 7)];

#pragma unroll
      for (int i = 0; i < 2; ++i) {
        acc[i][j2] = __builtin_amdgcn_wmma_f32_16x16x32_bf16(
            false, af[i], false, bf, (short)0, acc[i][j2], false, false);
      }
    }
    __syncthreads();
  }

  // ---- store: C/D layout -> VGPR r: lanes0-15 M=r, lanes16-31 M=8+r, N=l15
#pragma unroll
  for (int i = 0; i < 2; ++i) {
#pragma unroll
    for (int j2 = 0; j2 < 4; ++j2) {
      int gc  = colBase + wx * 64 + j2 * 16 + l15;
      int gr0 = rowBase + wy * 32 + i * 16 + hi * 8;
      float* p = D + (long long)gr0 * Cout + gc;
      if (fullTile) {
#pragma unroll
        for (int r = 0; r < 8; ++r) {
          *p = acc[i][j2][r];
          p += Cout;
        }
      } else {
#pragma unroll
        for (int r = 0; r < 8; ++r) {
          if (gr0 + r < N) *p = acc[i][j2][r];
          p += Cout;
        }
      }
    }
  }
}

// --------------------------- attention logits ------------------------------
// One wave (32 lanes) per node: s_src[i]=dot(H[i],a_src), s_dst likewise.

__global__ void logits_kernel(const float* __restrict__ H,
                              const float* __restrict__ a_src,
                              const float* __restrict__ a_dst,
                              float* __restrict__ ssrc, float* __restrict__ sdst,
                              int N, int C) {
  int node = blockIdx.x * (blockDim.x >> 5) + (threadIdx.x >> 5);
  int lane = threadIdx.x & 31;
  if (node >= N) return;
  float as = 0.f, ad = 0.f;
  const float* row = H + (long long)node * C;
  for (int c = lane; c < C; c += 32) {
    float h = row[c];
    as += h * a_src[c];
    ad += h * a_dst[c];
  }
#pragma unroll
  for (int off = 16; off; off >>= 1) {
    as += __shfl_xor(as, off);
    ad += __shfl_xor(ad, off);
  }
  if (lane == 0) {
    ssrc[node] = as;
    sdst[node] = ad;
  }
}

// ------------------------------ edge passes --------------------------------
// Order-preserving float->uint key so atomicMax(u32) implements float max.

__device__ __forceinline__ unsigned fkey(float f) {
  unsigned b = __float_as_uint(f);
  return (b & 0x80000000u) ? ~b : (b | 0x80000000u);
}
__device__ __forceinline__ float fkey_inv(unsigned k) {
  return (k & 0x80000000u) ? __uint_as_float(k ^ 0x80000000u)
                           : __uint_as_float(~k);
}

__global__ void edge_max_kernel(const int* __restrict__ ei,
                                const float* __restrict__ ssrc,
                                const float* __restrict__ sdst,
                                unsigned* __restrict__ mkey, int E) {
  int e = blockIdx.x * blockDim.x + threadIdx.x;
  if (e >= E) return;
  int s = ei[e], d = ei[E + e];
  float v = leaky(ssrc[s] + sdst[d]);
  atomicMax(&mkey[d], fkey(v));
}

__global__ void edge_exp_kernel(const int* __restrict__ ei,
                                const float* __restrict__ ssrc,
                                const float* __restrict__ sdst,
                                const unsigned* __restrict__ mkey,
                                float* __restrict__ ex, float* __restrict__ z,
                                int E) {
  int e = blockIdx.x * blockDim.x + threadIdx.x;
  if (e >= E) return;
  int s = ei[e], d = ei[E + e];
  float v = leaky(ssrc[s] + sdst[d]);
  float m = fkey_inv(mkey[d]);
  float ev = __expf(v - m);
  ex[e] = ev;
  atomicAdd(&z[d], ev);
}

// out[dst, c..c+3] += H[src, c..c+3] * alpha ; one thread = 4 channels
__global__ void scatter_kernel(const int* __restrict__ ei,
                               const float* __restrict__ H,
                               const float* __restrict__ ex,
                               const float* __restrict__ z,
                               float* __restrict__ out, int E, int C) {
  long long g = (long long)blockIdx.x * blockDim.x + threadIdx.x;
  int perEdge = C >> 2;
  long long total = (long long)E * perEdge;
  if (g >= total) return;
  int e  = (int)(g / perEdge);
  int c4 = (int)(g % perEdge) << 2;
  int s = ei[e], d = ei[E + e];
  float alpha = ex[e] / (z[d] + 1e-16f);
  float4 hv = *(const float4*)(H + (long long)s * C + c4);
  float* o = out + (long long)d * C + c4;
  atomicAdd(o + 0, hv.x * alpha);
  atomicAdd(o + 1, hv.y * alpha);
  atomicAdd(o + 2, hv.z * alpha);
  atomicAdd(o + 3, hv.w * alpha);
}

// --------------------------- epilogue kernels ------------------------------

__global__ void bias_init_kernel(float* __restrict__ out,
                                 const float* __restrict__ b, long long n, int C) {
  long long i = (long long)blockIdx.x * blockDim.x + threadIdx.x;
  long long stride = (long long)gridDim.x * blockDim.x;
  for (; i < n; i += stride) out[i] = b[(int)(i % C)];
}

__global__ void relu_bias_bf16_kernel(const float* __restrict__ acc,
                                      const float* __restrict__ b,
                                      __bf16* __restrict__ out, long long n, int C) {
  long long i = (long long)blockIdx.x * blockDim.x + threadIdx.x;
  long long stride = (long long)gridDim.x * blockDim.x;
  for (; i < n; i += stride) {
    float v = acc[i] + b[(int)(i % C)];
    out[i] = f2bf(v > 0.f ? v : 0.f);
  }
}

// ------------------------------- launcher ----------------------------------

extern "C" void kernel_launch(void* const* d_in, const int* in_sizes, int n_in,
                              void* d_out, int out_size, void* d_ws, size_t ws_size,
                              hipStream_t stream) {
  (void)in_sizes; (void)n_in; (void)out_size; (void)ws_size;
  const float* x        = (const float*)d_in[0];
  const int*   ei       = (const int*)d_in[1];   // [2,E]: src row then dst row
  const float* W1       = (const float*)d_in[2];
  const float* a_src1   = (const float*)d_in[3];
  const float* a_dst1   = (const float*)d_in[4];
  const float* b1       = (const float*)d_in[5];
  const float* W_mu     = (const float*)d_in[6];
  const float* a_src_mu = (const float*)d_in[7];
  const float* a_dst_mu = (const float*)d_in[8];
  const float* b_mu     = (const float*)d_in[9];
  const float* W_ls     = (const float*)d_in[10];
  const float* a_src_ls = (const float*)d_in[11];
  const float* a_dst_ls = (const float*)d_in[12];
  const float* b_ls     = (const float*)d_in[13];

  const int N = GAT_N, E = GAT_E;
  const int C_IN = 256, C_HID = 256, C_OUT = 128;

  // ---- workspace carve-out (~205 MB)
  char* ws = (char*)d_ws;
  size_t o = 0;
  auto carve = [&](size_t bytes) -> void* {
    void* p = ws + o;
    o = (o + bytes + 255) & ~(size_t)255;
    return p;
  };
  __bf16*   XBF   = (__bf16*)  carve((size_t)N * C_IN * 2);
  float*    H1    = (float*)   carve((size_t)N * C_HID * 4);
  float*    OUT1  = (float*)   carve((size_t)N * C_HID * 4);
  __bf16*   H1BF  = (__bf16*)  carve((size_t)N * C_HID * 2);
  float*    HMU   = (float*)   carve((size_t)N * C_OUT * 4);
  float*    HLS   = (float*)   carve((size_t)N * C_OUT * 4);
  __bf16*   W1T   = (__bf16*)  carve((size_t)C_IN * C_HID * 2);   // [C_HID,C_IN]
  __bf16*   WMUT  = (__bf16*)  carve((size_t)C_HID * C_OUT * 2);  // [C_OUT,C_HID]
  __bf16*   WLST  = (__bf16*)  carve((size_t)C_HID * C_OUT * 2);  // [C_OUT,C_HID]
  float*    SSRC  = (float*)   carve((size_t)N * 4);
  float*    SDST  = (float*)   carve((size_t)N * 4);
  unsigned* MKEY  = (unsigned*)carve((size_t)N * 4);
  float*    Z     = (float*)   carve((size_t)N * 4);
  float*    EX    = (float*)   carve((size_t)E * 4);

  const int eb = (E + 255) / 256;
  const int nodeBlocks = (N + 7) / 8;  // 8 waves/block, wave per node

  // ---- bf16 conversions (weights transposed to [Cout,K])
  f32_to_bf16_kernel<<<2048, 256, 0, stream>>>(x, XBF, (long long)N * C_IN);
  f32_to_bf16_t_kernel<<<64, 256, 0, stream>>>(W1, W1T, C_IN, C_HID);
  f32_to_bf16_t_kernel<<<64, 256, 0, stream>>>(W_mu, WMUT, C_HID, C_OUT);
  f32_to_bf16_t_kernel<<<64, 256, 0, stream>>>(W_ls, WLST, C_HID, C_OUT);

  // ================= layer 1 =================
  {
    dim3 g((N + 127) / 128, C_HID / 128);
    gemm_bf16_wmma<<<g, 256, 0, stream>>>(XBF, W1T, H1, N, C_IN, C_HID);
  }
  logits_kernel<<<nodeBlocks, 256, 0, stream>>>(H1, a_src1, a_dst1, SSRC, SDST, N, C_HID);
  hipMemsetAsync(MKEY, 0, (size_t)N * 4, stream);
  hipMemsetAsync(Z, 0, (size_t)N * 4, stream);
  edge_max_kernel<<<eb, 256, 0, stream>>>(ei, SSRC, SDST, MKEY, E);
  edge_exp_kernel<<<eb, 256, 0, stream>>>(ei, SSRC, SDST, MKEY, EX, Z, E);
  hipMemsetAsync(OUT1, 0, (size_t)N * C_HID * 4, stream);
  {
    long long total = (long long)E * (C_HID / 4);
    scatter_kernel<<<(unsigned)((total + 255) / 256), 256, 0, stream>>>(
        ei, H1, EX, Z, OUT1, E, C_HID);
  }
  relu_bias_bf16_kernel<<<2048, 256, 0, stream>>>(OUT1, b1, H1BF, (long long)N * C_HID, C_HID);

  // ================= mu head =================
  float* outMu = (float*)d_out;
  {
    dim3 g((N + 127) / 128, C_OUT / 128);
    gemm_bf16_wmma<<<g, 256, 0, stream>>>(H1BF, WMUT, HMU, N, C_HID, C_OUT);
  }
  logits_kernel<<<nodeBlocks, 256, 0, stream>>>(HMU, a_src_mu, a_dst_mu, SSRC, SDST, N, C_OUT);
  hipMemsetAsync(MKEY, 0, (size_t)N * 4, stream);
  hipMemsetAsync(Z, 0, (size_t)N * 4, stream);
  edge_max_kernel<<<eb, 256, 0, stream>>>(ei, SSRC, SDST, MKEY, E);
  edge_exp_kernel<<<eb, 256, 0, stream>>>(ei, SSRC, SDST, MKEY, EX, Z, E);
  bias_init_kernel<<<2048, 256, 0, stream>>>(outMu, b_mu, (long long)N * C_OUT, C_OUT);
  {
    long long total = (long long)E * (C_OUT / 4);
    scatter_kernel<<<(unsigned)((total + 255) / 256), 256, 0, stream>>>(
        ei, HMU, EX, Z, outMu, E, C_OUT);
  }

  // ================= logstd head =================
  float* outLs = outMu + (size_t)N * C_OUT;
  {
    dim3 g((N + 127) / 128, C_OUT / 128);
    gemm_bf16_wmma<<<g, 256, 0, stream>>>(H1BF, WLST, HLS, N, C_HID, C_OUT);
  }
  logits_kernel<<<nodeBlocks, 256, 0, stream>>>(HLS, a_src_ls, a_dst_ls, SSRC, SDST, N, C_OUT);
  hipMemsetAsync(MKEY, 0, (size_t)N * 4, stream);
  hipMemsetAsync(Z, 0, (size_t)N * 4, stream);
  edge_max_kernel<<<eb, 256, 0, stream>>>(ei, SSRC, SDST, MKEY, E);
  edge_exp_kernel<<<eb, 256, 0, stream>>>(ei, SSRC, SDST, MKEY, EX, Z, E);
  bias_init_kernel<<<2048, 256, 0, stream>>>(outLs, b_ls, (long long)N * C_OUT, C_OUT);
  {
    long long total = (long long)E * (C_OUT / 4);
    scatter_kernel<<<(unsigned)((total + 255) / 256), 256, 0, stream>>>(
        ei, HLS, EX, Z, outLs, E, C_OUT);
  }
}
